// GraphTransformer_23648089932024
// MI455X (gfx1250) — matrix operand order repaired and checked by reading, not compiled, not run on hardware
//
#include <hip/hip_runtime.h>
#include <cmath>

// ---- problem constants (match reference) ----
constexpr int Bc = 16, Nc = 128, Dc = 512, Hc = 16, Lc = 6;
constexpr int FFNc = 2048, Sc = 129, MDc = 5, LAPc = 8, SVDc = 16, NCc = 10;
constexpr int Mrows = Bc * Sc; // 2064

typedef __bf16 bf16_t;
typedef __attribute__((ext_vector_type(16))) bf16_t v16bf;
typedef __attribute__((ext_vector_type(8)))  bf16_t v8bf;
typedef __attribute__((ext_vector_type(8)))  float  v8f;
typedef __attribute__((ext_vector_type(4)))  unsigned int u32x4;
typedef __attribute__((ext_vector_type(4)))  int i32x4;
typedef __attribute__((ext_vector_type(8)))  int i32x8;

static __device__ __forceinline__ v16bf cat16(v8bf lo, v8bf hi) {
  return __builtin_shufflevector(lo, hi, 0,1,2,3,4,5,6,7,8,9,10,11,12,13,14,15);
}
static __device__ __forceinline__ v8f wmma_bf16(v16bf a, v16bf b, v8f c) {
  return __builtin_amdgcn_wmma_f32_16x16x32_bf16(false, a, false, b, (short)0, c, false, false);
}
static __device__ __forceinline__ float gelu_exact(float v) {
  return 0.5f * v * (1.0f + erff(v * 0.70710678118654752f));
}

// ---------------------------------------------------------------------
// Tensor Data Mover: 2D tile load (global -> LDS), D# packed per ISA
// group0: [1:0]=count=1, [63:32]=lds_addr, [120:64]=global_addr, [127:126]=type 2
// group1: [17:16]=data_size, [79:48]=tensor_dim0, [111:80]=tensor_dim1,
//         [127:112]=tile_dim0, [143:128]=tile_dim1, [207:160]=tensor_dim0_stride
// ---------------------------------------------------------------------
static __device__ __forceinline__ void tdm_load_2d(
    unsigned lds_addr, const void* gaddr,
    unsigned tile_w_elems, unsigned tile_h,
    unsigned tensor_w, unsigned tensor_h,
    unsigned row_stride_elems, unsigned data_size_code /*1 = 2 bytes*/)
{
  unsigned long long ga = (unsigned long long)(uintptr_t)gaddr;
  u32x4 g0;
  g0[0] = 1u;                                         // count = 1 (user mode)
  g0[1] = lds_addr;                                   // lds_addr
  g0[2] = (unsigned)(ga & 0xffffffffu);               // global_addr lo
  g0[3] = (unsigned)((ga >> 32) & 0x01ffffffu) | 0x80000000u; // ga hi | type=2
  i32x8 g1;
  g1[0] = (int)(data_size_code << 16);                // data_size
  g1[1] = (int)((tensor_w & 0xffffu) << 16);          // tensor_dim0 lo16
  g1[2] = (int)(((tensor_w >> 16) & 0xffffu) | ((tensor_h & 0xffffu) << 16));
  g1[3] = (int)(((tensor_h >> 16) & 0xffffu) | ((tile_w_elems & 0xffffu) << 16));
  g1[4] = (int)(tile_h & 0xffffu);                    // tile_dim1 (tile_dim2=0)
  g1[5] = (int)row_stride_elems;                      // tensor_dim0_stride lo32
  g1[6] = 0;
  g1[7] = 0;
  i32x4 z4 = {0, 0, 0, 0};
#if defined(__clang_major__) && (__clang_major__ >= 23)
  i32x8 z8 = {0, 0, 0, 0, 0, 0, 0, 0};
  __builtin_amdgcn_tensor_load_to_lds(g0, g1, z4, z4, z8, 0);
#else
  __builtin_amdgcn_tensor_load_to_lds(g0, g1, z4, z4, 0);
#endif
}

// =====================================================================
// Attention-bias construction: gab[b,h,s,t]
// =====================================================================
__global__ __launch_bounds__(256) void k_build_bias(
    const float* __restrict__ attn_bias, const float* __restrict__ gt_vd,
    const int* __restrict__ spatial_pos, const float* __restrict__ spatial_emb,
    const int* __restrict__ edge_input,  const float* __restrict__ edge_emb,
    const float* __restrict__ edge_dis,  const float* __restrict__ pma_enc,
    const float* __restrict__ pma_w,     const float* __restrict__ pma_b,
    float* __restrict__ gab)
{
  int idx = blockIdx.x * blockDim.x + threadIdx.x;
  if (idx >= Bc * Sc * Sc) return;
  int t = idx % Sc;
  int s = (idx / Sc) % Sc;
  int b = idx / (Sc * Sc);

  float base = 2.0f * attn_bias[idx];
  float acc[Hc];
#pragma unroll
  for (int h = 0; h < Hc; ++h) acc[h] = base;

  if (s == 0 || t == 0) {
#pragma unroll
    for (int h = 0; h < Hc; ++h) acc[h] += gt_vd[h];
  }
  if (s > 0 && t > 0) {
    int n = s - 1, m = t - 1;
    size_t nm = ((size_t)(b * Nc + n)) * Nc + m;
    int sp = spatial_pos[nm];
#pragma unroll
    for (int h = 0; h < Hc; ++h) acc[h] += spatial_emb[sp * Hc + h];

    int spv = (sp == 0) ? 1 : sp;
    if (spv > 1) spv -= 1;
    if (spv > MDc) spv = MDc;
    float inv_sp = 1.0f / (float)spv;

    float etmp[Hc];
#pragma unroll
    for (int h = 0; h < Hc; ++h) etmp[h] = 0.0f;
    for (int d = 0; d < MDc; ++d) {
      int ei = edge_input[nm * MDc + d];
      for (int h2 = 0; h2 < Hc; ++h2) {
        float e = edge_emb[ei * Hc + h2];
        const float* wrow = edge_dis + (size_t)(d * Hc + h2) * Hc;
#pragma unroll
        for (int h = 0; h < Hc; ++h) etmp[h] += e * wrow[h];
      }
    }
#pragma unroll
    for (int h = 0; h < Hc; ++h) acc[h] += etmp[h] * inv_sp;

    float pv[MDc];
#pragma unroll
    for (int d = 0; d < MDc; ++d) pv[d] = pma_enc[nm * MDc + d];
#pragma unroll
    for (int h = 0; h < Hc; ++h) {
      float sum = pma_b[h];
#pragma unroll
      for (int d = 0; d < MDc; ++d) sum += pv[d] * pma_w[h * MDc + d];
      acc[h] += sum;
    }
  }
#pragma unroll
  for (int h = 0; h < Hc; ++h)
    gab[(((size_t)b * Hc + h) * Sc + s) * Sc + t] = acc[h];
}

// =====================================================================
// Node embedding + positional encoding init
// =====================================================================
__global__ __launch_bounds__(256) void k_init(
    const int* __restrict__ xin, const int* __restrict__ indeg,
    const float* __restrict__ atom_emb, const float* __restrict__ gtok,
    const float* __restrict__ ide, const float* __restrict__ ode,
    const float* __restrict__ lap_enc, const float* __restrict__ lap_w,
    const float* __restrict__ lap_b, const float* __restrict__ svd_enc,
    const float* __restrict__ svd_w, const float* __restrict__ svd_b,
    float* __restrict__ X, float* __restrict__ POS)
{
  int t = blockIdx.x * blockDim.x + threadIdx.x;
  if (t >= Bc * Sc * Dc) return;
  int d = t % Dc;
  int s = (t / Dc) % Sc;
  int b = t / (Dc * Sc);
  if (s == 0) { X[t] = gtok[d]; return; }
  int n = s - 1;
  float xv = 0.0f;
#pragma unroll
  for (int f = 0; f < 3; ++f)
    xv += atom_emb[(size_t)xin[(b * Nc + n) * 3 + f] * Dc + d];
  X[t] = xv;

  int deg = indeg[b * Nc + n];
  float pv = ide[(size_t)deg * Dc + d] + ode[(size_t)deg * Dc + d] + lap_b[d] + svd_b[d];
#pragma unroll
  for (int j = 0; j < LAPc; ++j)
    pv += lap_enc[(size_t)(b * Nc + n) * LAPc + j] * lap_w[(size_t)d * LAPc + j];
#pragma unroll
  for (int j = 0; j < SVDc; ++j)
    pv += svd_enc[(size_t)(b * Nc + n) * SVDc + j] * svd_w[(size_t)d * SVDc + j];
  POS[(size_t)(b * Nc + n) * Dc + d] = pv;
}

// =====================================================================
// LayerNorm (optionally adds positional enc into residual stream first)
// =====================================================================
__global__ __launch_bounds__(128) void k_ln(
    float* __restrict__ X, const float* __restrict__ POS,
    const float* __restrict__ g, const float* __restrict__ bb,
    bf16_t* __restrict__ Y, int addpos)
{
  int row = blockIdx.x;
  int b = row / Sc, s = row % Sc;
  int tid = threadIdx.x;
  __shared__ float ssum[128];
  __shared__ float ssq[128];

  float4 xv = *(float4*)(X + (size_t)row * Dc + tid * 4);
  if (addpos && s > 0) {
    float4 pv = *(const float4*)(POS + (size_t)(b * Nc + s - 1) * Dc + tid * 4);
    xv.x += pv.x; xv.y += pv.y; xv.z += pv.z; xv.w += pv.w;
    *(float4*)(X + (size_t)row * Dc + tid * 4) = xv;
  }
  ssum[tid] = xv.x + xv.y + xv.z + xv.w;
  ssq[tid]  = xv.x * xv.x + xv.y * xv.y + xv.z * xv.z + xv.w * xv.w;
  __syncthreads();
  for (int st = 64; st > 0; st >>= 1) {
    if (tid < st) { ssum[tid] += ssum[tid + st]; ssq[tid] += ssq[tid + st]; }
    __syncthreads();
  }
  float mean = ssum[0] * (1.0f / Dc);
  float var  = ssq[0] * (1.0f / Dc) - mean * mean;
  float rstd = rsqrtf(var + 1e-5f);
  int d = tid * 4;
  bf16_t* yp = Y + (size_t)row * Dc + d;
  yp[0] = (bf16_t)((xv.x - mean) * rstd * g[d + 0] + bb[d + 0]);
  yp[1] = (bf16_t)((xv.y - mean) * rstd * g[d + 1] + bb[d + 1]);
  yp[2] = (bf16_t)((xv.z - mean) * rstd * g[d + 2] + bb[d + 2]);
  yp[3] = (bf16_t)((xv.w - mean) * rstd * g[d + 3] + bb[d + 3]);
}

// =====================================================================
// WMMA GEMM: Out = epilogue( A(MxK bf16) @ W(NxK f32)^T + bias )
// Workgroup tile 32x128, 4 waves; each wave owns a 32x32 output block:
// 2 A-fragments x 2 B-fragments -> 4 WMMAs per K-step with operand reuse.
// mode bit0: output bf16 (else f32 residual add), bit1: exact GELU.
// =====================================================================
__global__ __launch_bounds__(128) void k_gemm(
    const bf16_t* __restrict__ A, const float* __restrict__ W,
    const float* __restrict__ bias, void* __restrict__ Out,
    int M, int N, int K, int mode, float alpha)
{
  __shared__ bf16_t As[32 * 32];   // 2 KB
  __shared__ bf16_t Bs[128 * 32];  // 8 KB, Bt[n][k]

  int tid = threadIdx.x;
  int wave = tid >> 5, lane = tid & 31;
  int hl = lane >> 4, nl = lane & 15;
  int m0 = blockIdx.x * 32, n0 = blockIdx.y * 128;
  int nbase = wave * 32;

  v8f acc00 = {0.f,0.f,0.f,0.f,0.f,0.f,0.f,0.f};
  v8f acc01 = acc00, acc10 = acc00, acc11 = acc00;

  for (int k0 = 0; k0 < K; k0 += 32) {
    { // stage A tile 32x32: 8 bf16 (16B) per thread
      int idx = tid * 8;
      int r = idx >> 5, c = idx & 31;
      int rg = m0 + r; if (rg > M - 1) rg = M - 1;
      *(uint4*)&As[idx] = *(const uint4*)(A + (size_t)rg * K + k0 + c);
    }
    { // stage W^T tile 128x32 (f32 -> bf16), one row per thread
      const float* src = W + (size_t)(n0 + tid) * K + k0;
      bf16_t* dst = &Bs[tid * 32];
#pragma unroll
      for (int i = 0; i < 32; i += 4) {
        float4 f = *(const float4*)(src + i);
        dst[i + 0] = (bf16_t)f.x; dst[i + 1] = (bf16_t)f.y;
        dst[i + 2] = (bf16_t)f.z; dst[i + 3] = (bf16_t)f.w;
      }
    }
    __syncthreads();
    if (k0 + 32 < K) { // next K-slice into cache (global_prefetch_b8)
      __builtin_prefetch(A + (size_t)(m0 + (tid >> 2)) * K + k0 + 32, 0, 1);
      __builtin_prefetch(W + (size_t)(n0 + tid) * K + k0 + 32, 0, 1);
    }
    const bf16_t* ap0 = &As[nl * 32 + hl * 8];
    const bf16_t* ap1 = &As[(16 + nl) * 32 + hl * 8];
    v16bf a0 = cat16(*(const v8bf*)ap0, *(const v8bf*)(ap0 + 16));
    v16bf a1 = cat16(*(const v8bf*)ap1, *(const v8bf*)(ap1 + 16));
    const bf16_t* bp0 = &Bs[(nbase + nl) * 32 + hl * 8];
    const bf16_t* bp1 = &Bs[(nbase + 16 + nl) * 32 + hl * 8];
    v16bf b0 = cat16(*(const v8bf*)bp0, *(const v8bf*)(bp0 + 16));
    v16bf b1 = cat16(*(const v8bf*)bp1, *(const v8bf*)(bp1 + 16));
    acc00 = wmma_bf16(a0, b0, acc00);
    acc01 = wmma_bf16(a0, b1, acc01);
    acc10 = wmma_bf16(a1, b0, acc10);
    acc11 = wmma_bf16(a1, b1, acc11);
    __syncthreads();
  }

  int col0 = n0 + nbase + nl;
  int col1 = col0 + 16;
  float bc0 = bias ? bias[col0] : 0.0f;
  float bc1 = bias ? bias[col1] : 0.0f;
#pragma unroll
  for (int r = 0; r < 8; ++r) {
    int rowA = m0 + r + hl * 8;      // acc0x rows
    int rowB = rowA + 16;            // acc1x rows
    float v00 = (acc00[r] + bc0) * alpha;
    float v01 = (acc01[r] + bc1) * alpha;
    float v10 = (acc10[r] + bc0) * alpha;
    float v11 = (acc11[r] + bc1) * alpha;
    if (mode & 2) { v00 = gelu_exact(v00); v01 = gelu_exact(v01);
                    v10 = gelu_exact(v10); v11 = gelu_exact(v11); }
    if (mode & 1) {
      bf16_t* O = (bf16_t*)Out;
      if (rowA < M) { O[(size_t)rowA * N + col0] = (bf16_t)v00;
                      O[(size_t)rowA * N + col1] = (bf16_t)v01; }
      if (rowB < M) { O[(size_t)rowB * N + col0] = (bf16_t)v10;
                      O[(size_t)rowB * N + col1] = (bf16_t)v11; }
    } else {
      float* O = (float*)Out;
      if (rowA < M) { O[(size_t)rowA * N + col0] += v00;
                      O[(size_t)rowA * N + col1] += v01; }
      if (rowB < M) { O[(size_t)rowB * N + col0] += v10;
                      O[(size_t)rowB * N + col1] += v11; }
    }
  }
}

// =====================================================================
// Fused attention, one wave per (query-tile, head, batch).
// TDM stages the whole (b,h) K-panel and V-panel into LDS (async tensor
// DMA, OOB rows beyond S return zero), then: scores = QK^T + gab ->
// softmax -> P@V, all WMMA. Keys padded with -inf bias => P = 0.
// =====================================================================
__global__ __launch_bounds__(32) void k_attn(
    const bf16_t* __restrict__ q, const bf16_t* __restrict__ k,
    const bf16_t* __restrict__ v, const float* __restrict__ gab,
    bf16_t* __restrict__ o)
{
  const int lane = threadIdx.x;
  const int hl = lane >> 4, nl = lane & 15;
  const int s0 = blockIdx.x * 16;
  const int h = blockIdx.y, b = blockIdx.z;

  __shared__ float  sc[16 * 144];     // score tile row
  __shared__ bf16_t Pt[16 * 160];     // softmax probs (zero-padded)
  __shared__ bf16_t Kpan[144 * 32];   // K panel for this (b,h)
  __shared__ bf16_t Vpan[160 * 32];   // V panel for this (b,h)
  __shared__ bf16_t Vt[32 * 32];      // transposed V tile Vt[d][t]

  // --- async tensor DMA: K and V panels (rows >= S read as zero) ---
  tdm_load_2d((unsigned)(uintptr_t)&Kpan[0],
              k + ((size_t)(b * Sc)) * Dc + h * 32,
              /*tile_w=*/32, /*tile_h=*/144, /*tensor_w=*/Dc, /*tensor_h=*/Sc,
              /*stride=*/Dc, /*dsize=2B*/1);
  tdm_load_2d((unsigned)(uintptr_t)&Vpan[0],
              v + ((size_t)(b * Sc)) * Dc + h * 32,
              32, 160, Dc, Sc, Dc, 1);

  // Q fragment (A layout): lane holds row s0+nl, K-chunks of dh
  int srow = s0 + nl; if (srow > Sc - 1) srow = Sc - 1;
  const bf16_t* qp = q + ((size_t)(b * Sc + srow)) * Dc + h * 32 + hl * 8;
  v16bf qf = cat16(*(const v8bf*)qp, *(const v8bf*)(qp + 16));

  __builtin_amdgcn_s_wait_tensorcnt(0);
  __syncthreads();

  for (int t0 = 0; t0 < 144; t0 += 16) {
    const bf16_t* bp = &Kpan[(t0 + nl) * 32 + hl * 8];
    v16bf kf = cat16(*(const v8bf*)bp, *(const v8bf*)(bp + 16));
    v8f scf = {0.f,0.f,0.f,0.f,0.f,0.f,0.f,0.f};
    scf = wmma_bf16(qf, kf, scf);
#pragma unroll
    for (int r = 0; r < 8; ++r) {
      int m = r + hl * 8;
      int sg = s0 + m, tgl = t0 + nl;
      float val;
      if (tgl < Sc) {
        int si = sg < Sc ? sg : Sc - 1;
        val = scf[r] + gab[(((size_t)b * Hc + h) * Sc + si) * Sc + tgl];
      } else {
        val = -1e30f;
      }
      sc[m * 144 + t0 + nl] = val;
    }
  }
  __syncthreads();

  // softmax per row; lanes 0..15 each own one row
  if (lane < 16) {
    float mx = -1e30f;
    for (int j = 0; j < 144; ++j) mx = fmaxf(mx, sc[lane * 144 + j]);
    float sum = 0.f;
    for (int j = 0; j < 144; ++j) {
      float e = __expf(sc[lane * 144 + j] - mx);
      sc[lane * 144 + j] = e;
      sum += e;
    }
    float inv = 1.0f / sum;
    for (int j = 0; j < 144; ++j) Pt[lane * 160 + j] = (bf16_t)(sc[lane * 144 + j] * inv);
    for (int j = 144; j < 160; ++j) Pt[lane * 160 + j] = (bf16_t)0.0f;
  }
  __syncthreads();

  v8f acc0 = {0.f,0.f,0.f,0.f,0.f,0.f,0.f,0.f};
  v8f acc1 = acc0;
  for (int kt = 0; kt < 160; kt += 32) {
    // transpose V tile inside LDS: Vt[d][t] = Vpan[t][d]
    for (int tt = 0; tt < 32; ++tt)
      Vt[lane * 32 + tt] = Vpan[(kt + tt) * 32 + lane];
    __syncthreads();
    const bf16_t* pp = &Pt[nl * 160 + kt + hl * 8];
    v16bf pf = cat16(*(const v8bf*)pp, *(const v8bf*)(pp + 16));
    const bf16_t* v0p = &Vt[nl * 32 + hl * 8];
    v16bf v0f = cat16(*(const v8bf*)v0p, *(const v8bf*)(v0p + 16));
    const bf16_t* v1p = &Vt[(16 + nl) * 32 + hl * 8];
    v16bf v1f = cat16(*(const v8bf*)v1p, *(const v8bf*)(v1p + 16));
    acc0 = wmma_bf16(pf, v0f, acc0);
    acc1 = wmma_bf16(pf, v1f, acc1);
    __syncthreads();
  }

#pragma unroll
  for (int r = 0; r < 8; ++r) {
    int sg = s0 + r + hl * 8;
    if (sg < Sc) {
      size_t baseo = ((size_t)(b * Sc + sg)) * Dc + h * 32;
      o[baseo + nl]      = (bf16_t)acc0[r];
      o[baseo + 16 + nl] = (bf16_t)acc1[r];
    }
  }
}

// =====================================================================
// Final LN (token 0 only) + classifier head
// =====================================================================
__global__ __launch_bounds__(128) void k_final(
    const float* __restrict__ X, const float* __restrict__ g,
    const float* __restrict__ bb, const float* __restrict__ ow,
    const float* __restrict__ ob, float* __restrict__ out)
{
  int b = blockIdx.x, tid = threadIdx.x;
  __shared__ float y[Dc];
  __shared__ float ssum[128];
  __shared__ float ssq[128];

  float4 xv = *(const float4*)(X + ((size_t)b * Sc) * Dc + tid * 4);
  ssum[tid] = xv.x + xv.y + xv.z + xv.w;
  ssq[tid]  = xv.x * xv.x + xv.y * xv.y + xv.z * xv.z + xv.w * xv.w;
  __syncthreads();
  for (int st = 64; st > 0; st >>= 1) {
    if (tid < st) { ssum[tid] += ssum[tid + st]; ssq[tid] += ssq[tid + st]; }
    __syncthreads();
  }
  float mean = ssum[0] * (1.0f / Dc);
  float var  = ssq[0] * (1.0f / Dc) - mean * mean;
  float rstd = rsqrtf(var + 1e-5f);
  int d = tid * 4;
  y[d + 0] = (xv.x - mean) * rstd * g[d + 0] + bb[d + 0];
  y[d + 1] = (xv.y - mean) * rstd * g[d + 1] + bb[d + 1];
  y[d + 2] = (xv.z - mean) * rstd * g[d + 2] + bb[d + 2];
  y[d + 3] = (xv.w - mean) * rstd * g[d + 3] + bb[d + 3];
  __syncthreads();
  if (tid < NCc) {
    float acc = ob[tid];
    for (int j = 0; j < Dc; ++j) acc += y[j] * ow[(size_t)tid * Dc + j];
    out[b * NCc + tid] = acc;
  }
}

// =====================================================================
extern "C" void kernel_launch(void* const* d_in, const int* in_sizes, int n_in,
                              void* d_out, int out_size, void* d_ws, size_t ws_size,
                              hipStream_t stream)
{
  (void)in_sizes; (void)n_in; (void)out_size; (void)ws_size;

  const float* attn_bias   = (const float*)d_in[0];
  const float* lap_enc     = (const float*)d_in[1];
  const float* svd_enc     = (const float*)d_in[2];
  const float* pma_enc     = (const float*)d_in[3];
  const float* atom_emb    = (const float*)d_in[4];
  const float* edge_emb    = (const float*)d_in[5];
  const float* edge_dis    = (const float*)d_in[6];
  const float* spatial_emb = (const float*)d_in[7];
  const float* in_deg_emb  = (const float*)d_in[8];
  const float* out_deg_emb = (const float*)d_in[9];
  const float* lap_w       = (const float*)d_in[10];
  const float* lap_b       = (const float*)d_in[11];
  const float* svd_w       = (const float*)d_in[12];
  const float* svd_b       = (const float*)d_in[13];
  const float* pma_w       = (const float*)d_in[14];
  const float* pma_b       = (const float*)d_in[15];
  const float* graph_token = (const float*)d_in[16];
  const float* gt_vd       = (const float*)d_in[17];
  const float* ln1_g       = (const float*)d_in[18];
  const float* ln1_b       = (const float*)d_in[19];
  const float* wq          = (const float*)d_in[20];
  const float* bq          = (const float*)d_in[21];
  const float* wk          = (const float*)d_in[22];
  const float* bk          = (const float*)d_in[23];
  const float* wv          = (const float*)d_in[24];
  const float* bv          = (const float*)d_in[25];
  const float* wo          = (const float*)d_in[26];
  const float* bo          = (const float*)d_in[27];
  const float* ln2_g       = (const float*)d_in[28];
  const float* ln2_b       = (const float*)d_in[29];
  const float* w1          = (const float*)d_in[30];
  const float* b1          = (const float*)d_in[31];
  const float* w2          = (const float*)d_in[32];
  const float* b2          = (const float*)d_in[33];
  const float* fln_g       = (const float*)d_in[34];
  const float* fln_b       = (const float*)d_in[35];
  const float* out_w       = (const float*)d_in[36];
  const float* out_b       = (const float*)d_in[37];
  const int* xin           = (const int*)d_in[38];
  const int* spatial_pos   = (const int*)d_in[39];
  const int* in_degree     = (const int*)d_in[40];
  const int* edge_input    = (const int*)d_in[41];

  char* ws = (char*)d_ws;
  size_t off = 0;
  auto walloc = [&](size_t bytes) -> void* {
    void* p = ws + off;
    off += (bytes + 255) & ~(size_t)255;
    return p;
  };
  float*  GAB = (float*)walloc((size_t)Bc * Hc * Sc * Sc * 4);
  float*  X   = (float*)walloc((size_t)Mrows * Dc * 4);
  float*  POS = (float*)walloc((size_t)Bc * Nc * Dc * 4);
  bf16_t* Y   = (bf16_t*)walloc((size_t)Mrows * Dc * 2);
  bf16_t* Q   = (bf16_t*)walloc((size_t)Mrows * Dc * 2);
  bf16_t* Kb  = (bf16_t*)walloc((size_t)Mrows * Dc * 2);
  bf16_t* Vb  = (bf16_t*)walloc((size_t)Mrows * Dc * 2);
  bf16_t* O   = (bf16_t*)walloc((size_t)Mrows * Dc * 2);
  bf16_t* Hb  = (bf16_t*)walloc((size_t)Mrows * FFNc * 2);

  k_build_bias<<<(Bc * Sc * Sc + 255) / 256, 256, 0, stream>>>(
      attn_bias, gt_vd, spatial_pos, spatial_emb, edge_input, edge_emb,
      edge_dis, pma_enc, pma_w, pma_b, GAB);
  k_init<<<(Bc * Sc * Dc + 255) / 256, 256, 0, stream>>>(
      xin, in_degree, atom_emb, graph_token, in_deg_emb, out_deg_emb,
      lap_enc, lap_w, lap_b, svd_enc, svd_w, svd_b, X, POS);

  const float qscale = 0.17677669529663687f; // 1/sqrt(32)
  dim3 gProj((Mrows + 31) / 32, Dc / 128);    // 65 x 4
  dim3 gFFN1((Mrows + 31) / 32, FFNc / 128);  // 65 x 16

  for (int l = 0; l < Lc; ++l) {
    k_ln<<<Mrows, 128, 0, stream>>>(X, POS, ln1_g + l * Dc, ln1_b + l * Dc, Y, 1);
    k_gemm<<<gProj, 128, 0, stream>>>(Y, wq + (size_t)l * Dc * Dc, bq + l * Dc, Q,
                                      Mrows, Dc, Dc, 1, qscale);
    k_gemm<<<gProj, 128, 0, stream>>>(Y, wk + (size_t)l * Dc * Dc, bk + l * Dc, Kb,
                                      Mrows, Dc, Dc, 1, 1.0f);
    k_gemm<<<gProj, 128, 0, stream>>>(Y, wv + (size_t)l * Dc * Dc, bv + l * Dc, Vb,
                                      Mrows, Dc, Dc, 1, 1.0f);
    k_attn<<<dim3(9, Hc, Bc), 32, 0, stream>>>(Q, Kb, Vb, GAB, O);
    k_gemm<<<gProj, 128, 0, stream>>>(O, wo + (size_t)l * Dc * Dc, bo + l * Dc, X,
                                      Mrows, Dc, Dc, 0, 1.0f); // residual add
    k_ln<<<Mrows, 128, 0, stream>>>(X, nullptr, ln2_g + l * Dc, ln2_b + l * Dc, Y, 0);
    k_gemm<<<gFFN1, 128, 0, stream>>>(Y, w1 + (size_t)l * FFNc * Dc, b1 + l * FFNc, Hb,
                                      Mrows, FFNc, Dc, 3, 1.0f);  // GELU -> bf16
    k_gemm<<<gProj, 128, 0, stream>>>(Hb, w2 + (size_t)l * Dc * FFNc, b2 + l * Dc, X,
                                      Mrows, Dc, FFNc, 0, 1.0f); // residual add
  }
  k_final<<<Bc, 128, 0, stream>>>(X, fln_g, fln_b, out_w, out_b, (float*)d_out);
}
